// DotProductAttention_16355235463421
// MI455X (gfx1250) — compile-verified, matrix operand order checked by reading, not run on hardware
//
#include <hip/hip_runtime.h>
#include <math.h>

// CDNA5 / gfx1250: wave32, WMMA 16x16x32 bf16 (f32 accumulate), TDM async copy.
typedef __bf16 v16bf __attribute__((ext_vector_type(16)));
typedef float  v8f   __attribute__((ext_vector_type(8)));
typedef unsigned int u32x4 __attribute__((ext_vector_type(4)));
typedef int          i32x4 __attribute__((ext_vector_type(4)));
typedef int          i32x8 __attribute__((ext_vector_type(8)));

#define BATCH   4
#define S_LEN   4096
#define DMODEL  512   // DM == DQ == DK == DV == 512

__device__ __forceinline__ __bf16 f2bf(float x) { return (__bf16)x; }

// LDS byte offset of a __shared__ object: flat shared addresses carry the LDS
// offset in their low 32 bits (ISA 10.2 aperture mapping).
__device__ __forceinline__ unsigned int lds_off(const void* p) {
    return (unsigned int)(uintptr_t)p;
}

// ---------------------------------------------------------------------------
// TDM: async 2D tile load Global -> LDS (bf16 elements, data_size = 2 bytes).
// D# packed per CDNA5 ISA:
//   group0: count=1 | lds_addr<<32 | global_addr@[120:64] | type2@[127:126]
//   group1: data_size=1(2B) | tensor_dim0/1 | tile_dim0/1 | tensor_dim0_stride
// Tile is described as its own tensor (global_addr points at the tile start),
// so tensor dims == tile dims and only the dim0 stride carries the row pitch.
// ---------------------------------------------------------------------------
__device__ __forceinline__ void tdm_load_2d_bf16(unsigned int lds_addr,
                                                 const void* gptr,
                                                 unsigned int tile_d0,   // elems per row
                                                 unsigned int tile_d1,   // rows
                                                 unsigned long long stride0) // elems
{
    const unsigned long long ga = (unsigned long long)(uintptr_t)gptr;

    const unsigned long long g0q0 = 1ull | ((unsigned long long)lds_addr << 32);
    const unsigned long long g0q1 = (ga & ((1ull << 57) - 1)) | (2ull << 62);

    const unsigned long long g1q0 = (1ull << 16)                                     // data_size = 2B
                                  | ((unsigned long long)(tile_d0 & 0xFFFFu) << 48); // tensor_dim0[15:0]
    const unsigned long long g1q1 = ((unsigned long long)tile_d0 >> 16)              // tensor_dim0[31:16]
                                  | ((unsigned long long)tile_d1 << 16)              // tensor_dim1
                                  | ((unsigned long long)tile_d0 << 48);             // tile_dim0
    const unsigned long long g1q2 = (unsigned long long)tile_d1                      // tile_dim1
                                  | ((stride0 & 0xFFFFFFFFull) << 32);               // stride0[31:0]
    const unsigned long long g1q3 = (stride0 >> 32);                                 // stride0[47:32]

    u32x4 g0; i32x8 g1; i32x4 g2; i32x4 g3;
    g0[0] = (unsigned int)g0q0;  g0[1] = (unsigned int)(g0q0 >> 32);
    g0[2] = (unsigned int)g0q1;  g0[3] = (unsigned int)(g0q1 >> 32);
    g1[0] = (int)(unsigned int)g1q0;  g1[1] = (int)(unsigned int)(g1q0 >> 32);
    g1[2] = (int)(unsigned int)g1q1;  g1[3] = (int)(unsigned int)(g1q1 >> 32);
    g1[4] = (int)(unsigned int)g1q2;  g1[5] = (int)(unsigned int)(g1q2 >> 32);
    g1[6] = (int)(unsigned int)g1q3;  g1[7] = (int)(unsigned int)(g1q3 >> 32);
    g2[0] = g2[1] = g2[2] = g2[3] = 0;   // 2D tensor: groups 2/3 unused
    g3[0] = g3[1] = g3[2] = g3[3] = 0;

#if defined(__clang_major__) && (__clang_major__ >= 23)
    i32x8 g4;
    g4[0]=g4[1]=g4[2]=g4[3]=g4[4]=g4[5]=g4[6]=g4[7]=0;
    __builtin_amdgcn_tensor_load_to_lds(g0, g1, g2, g3, g4, 0);
#else
    __builtin_amdgcn_tensor_load_to_lds(g0, g1, g2, g3, 0);
#endif
}

// ---------------------------------------------------------------------------
// Stage 1: C[M=16384][512] = A[M][512](f32) * W[512][512]^T + bias, bf16 out.
// transposed==0: out[row*512+col] (Q, K);  transposed==1: Vt laid out [B][512][S].
// Grid (128, 4), block 256 (8 waves); wave computes a 16x128 C tile.
// The 128x512 W tile is staged ONCE per block into LDS as bf16 (128 KB), so
// B-operands come from ds_load and W is read/converted once per block instead
// of once per wave (8x less L2 traffic + no load-cvt chain in front of WMMA).
// ---------------------------------------------------------------------------
__global__ __launch_bounds__(256)
void proj_gemm_bf16(const float* __restrict__ A,
                    const float* __restrict__ W,
                    const float* __restrict__ bias,
                    __bf16* __restrict__ out,
                    int transposed)
{
    __shared__ __align__(128) __bf16 wlds[128 * DMODEL];   // 128 KB

    const int tid  = threadIdx.x;
    const int lane = tid & 31;
    const int wave = tid >> 5;
    const int half = lane >> 4;
    const int l16  = lane & 15;
    const int row0 = blockIdx.x * 128 + wave * 16;
    const int col0 = blockIdx.y * 128;

    // Cooperative W-tile stage: 128 rows x 512, 256 elems/thread, bf16 convert.
    {
        const int wrow  = tid >> 1;            // 0..127
        const int dbase = (tid & 1) * 256;
        const float* wp = W + (size_t)(col0 + wrow) * DMODEL + dbase;
        __bf16* dst = &wlds[wrow * DMODEL + dbase];
#pragma unroll
        for (int j = 0; j < 16; ++j) {
            v16bf t;
#pragma unroll
            for (int e = 0; e < 16; ++e) t[e] = f2bf(wp[j * 16 + e]);
            *(v16bf*)&dst[j * 16] = t;
        }
    }
    __syncthreads();

    v8f acc[8];
#pragma unroll
    for (int nt = 0; nt < 8; ++nt) acc[nt] = (v8f)0.0f;

#pragma unroll 4
    for (int kk = 0; kk < DMODEL / 32; ++kk) {
        // A operand: 16x32 bf16 chunk (lane l16 holds row l16, halves split K)
        const float* ap = A + (size_t)(row0 + l16) * DMODEL + kk * 32 + half * 16;
        v16bf a;
#pragma unroll
        for (int e = 0; e < 16; ++e) a[e] = f2bf(ap[e]);

#pragma unroll
        for (int nt = 0; nt < 8; ++nt) {
            // B operand (W^T chunk 32x16) from LDS: lane n reads W row contiguous
            v16bf b = *(const v16bf*)&wlds[(size_t)(nt * 16 + l16) * DMODEL + kk * 32 + half * 16];
            acc[nt] = __builtin_amdgcn_wmma_f32_16x16x32_bf16(
                false, a, false, b, (short)0, acc[nt], false, false);
        }
    }

#pragma unroll
    for (int nt = 0; nt < 8; ++nt) {
        const int   col  = col0 + nt * 16 + l16;
        const float bcol = bias[col];
#pragma unroll
        for (int i = 0; i < 8; ++i) {
            const int row = row0 + i + 8 * half;   // WMMA C layout: VGPR i -> M = i + 8*half
            const float v = acc[nt][i] + bcol;
            size_t idx;
            if (!transposed) {
                idx = (size_t)row * DMODEL + col;
            } else {
                const int b = row >> 12;           // row / 4096
                const int s = row & (S_LEN - 1);
                idx = ((size_t)b * DMODEL + col) * S_LEN + s;
            }
            out[idx] = f2bf(v);
        }
    }
}

// ---------------------------------------------------------------------------
// Stage 2: flash attention, tril-to-ZERO mask semantics (masked scores are 0,
// softmax over ALL keys). Grid (S/64, B), block 128 (4 waves); each wave owns
// 16 q-rows x DV=512. K/V tiles are DMA'd Global->LDS by the Tensor Data
// Mover (double-buffered, TENSORcnt-tracked); WMMA operands come from LDS.
// ---------------------------------------------------------------------------
__global__ __launch_bounds__(128)
void attn_fwd_bf16(const __bf16* __restrict__ Q,   // [B*S][512]
                   const __bf16* __restrict__ K,   // [B*S][512]
                   const __bf16* __restrict__ Vt,  // [B][512][S]
                   float* __restrict__ O)          // [B*S][512]
{
    __shared__ __align__(128) __bf16 kbuf[2][32 * DMODEL];   // 2 x 32KB
    __shared__ __align__(128) __bf16 vbuf[2][DMODEL * 32];   // 2 x 32KB
    __shared__ __align__(64)  __bf16 pscr[4][16 * 32];       // wave-private P

    const int lane = threadIdx.x & 31;
    const int wave = threadIdx.x >> 5;
    const int half = lane >> 4;
    const int l16  = lane & 15;
    const int b    = blockIdx.y;
    const int q0   = blockIdx.x * 64 + wave * 16;      // q row within sequence
    const float scale = 0.044194173824159216f;         // 1/sqrt(512)

    const __bf16* Kb = K  + (size_t)b * S_LEN * DMODEL;
    const __bf16* Vb = Vt + (size_t)b * DMODEL * S_LEN;

    // Q strip 16x512 in registers (A-operand chunks).
    v16bf q[16];
#pragma unroll
    for (int c = 0; c < 16; ++c)
        q[c] = *(const v16bf*)(Q + (size_t)(b * S_LEN + q0 + l16) * DMODEL + c * 32 + half * 16);

    v8f o[32];
#pragma unroll
    for (int nt = 0; nt < 32; ++nt) o[nt] = (v8f)0.0f;
    float mrow[8], lrow[8];
#pragma unroll
    for (int i = 0; i < 8; ++i) { mrow[i] = -__builtin_inff(); lrow[i] = 0.0f; }

    const int NKB = S_LEN / 32;

    // Prologue: wave 0 kicks off the TDM pair for key-block 0.
    if (wave == 0) {
        tdm_load_2d_bf16(lds_off(&kbuf[0][0]), Kb, DMODEL, 32, DMODEL);
        tdm_load_2d_bf16(lds_off(&vbuf[0][0]), Vb, 32, DMODEL, S_LEN);
    }

    for (int kb = 0; kb < NKB; ++kb) {
        if (wave == 0) {
            if (kb + 1 < NKB) {   // issue next pair, then retire the current pair
                const int nb = kb + 1;
                tdm_load_2d_bf16(lds_off(&kbuf[nb & 1][0]),
                                 Kb + (size_t)nb * 32 * DMODEL, DMODEL, 32, DMODEL);
                tdm_load_2d_bf16(lds_off(&vbuf[nb & 1][0]),
                                 Vb + (size_t)nb * 32, 32, DMODEL, S_LEN);
                __builtin_amdgcn_s_wait_tensorcnt(2);
            } else {
                __builtin_amdgcn_s_wait_tensorcnt(0);
            }
        }
        __syncthreads();   // publish LDS tiles to all waves

        const __bf16* kt = kbuf[kb & 1];
        const __bf16* vt = vbuf[kb & 1];

        // ---- S = Q * K^T for a 16x32 key strip (two 16x16 C tiles) ----
        v8f s0 = (v8f)0.0f, s1 = (v8f)0.0f;
#pragma unroll
        for (int c = 0; c < 16; ++c) {
            v16bf k0 = *(const v16bf*)&kt[(size_t)l16        * DMODEL + c * 32 + half * 16];
            v16bf k1 = *(const v16bf*)&kt[(size_t)(16 + l16) * DMODEL + c * 32 + half * 16];
            s0 = __builtin_amdgcn_wmma_f32_16x16x32_bf16(false, q[c], false, k0, (short)0, s0, false, false);
            s1 = __builtin_amdgcn_wmma_f32_16x16x32_bf16(false, q[c], false, k1, (short)0, s1, false, false);
        }

        // ---- mask-to-zero + online softmax (per-row stats in lane halves) ----
#pragma unroll
        for (int i = 0; i < 8; ++i) {
            const int qrow = q0 + i + 8 * half;
            const int k0i  = kb * 32 + l16;
            const int k1i  = k0i + 16;
            const float x0 = (k0i <= qrow) ? s0[i] * scale : 0.0f;
            const float x1 = (k1i <= qrow) ? s1[i] * scale : 0.0f;

            float rm = fmaxf(x0, x1);
            rm = fmaxf(rm, __shfl_xor(rm, 1, 32));
            rm = fmaxf(rm, __shfl_xor(rm, 2, 32));
            rm = fmaxf(rm, __shfl_xor(rm, 4, 32));
            rm = fmaxf(rm, __shfl_xor(rm, 8, 32));
            const float mnew = fmaxf(mrow[i], rm);

            const float p0 = __expf(x0 - mnew);
            const float p1 = __expf(x1 - mnew);
            float rs = p0 + p1;
            rs += __shfl_xor(rs, 1, 32);
            rs += __shfl_xor(rs, 2, 32);
            rs += __shfl_xor(rs, 4, 32);
            rs += __shfl_xor(rs, 8, 32);

            const float alpha = __expf(mrow[i] - mnew);
            lrow[i] = lrow[i] * alpha + rs;
            mrow[i] = mnew;
#pragma unroll
            for (int nt = 0; nt < 32; ++nt) o[nt][i] *= alpha;

            pscr[wave][(i + 8 * half) * 32 +      l16] = f2bf(p0);
            pscr[wave][(i + 8 * half) * 32 + 16 + l16] = f2bf(p1);
        }

        // re-read P as A-operand (16x32); wave-private LDS, DS ops are in-order
        v16bf p = *(const v16bf*)&pscr[wave][l16 * 32 + half * 16];

        // ---- O += P * V  (Vt tile rows are contiguous in key dim) ----
#pragma unroll
        for (int nt = 0; nt < 32; ++nt) {
            v16bf vv = *(const v16bf*)&vt[(size_t)(nt * 16 + l16) * 32 + half * 16];
            o[nt] = __builtin_amdgcn_wmma_f32_16x16x32_bf16(false, p, false, vv, (short)0, o[nt], false, false);
        }

        __syncthreads();   // all reads of buf[kb&1] done before TDM reuses it
    }

    // ---- epilogue: O /= l, store f32 ----
#pragma unroll
    for (int i = 0; i < 8; ++i) {
        const float inv = 1.0f / lrow[i];
        const size_t rbase = (size_t)(b * S_LEN + q0 + i + 8 * half) * DMODEL;
#pragma unroll
        for (int nt = 0; nt < 32; ++nt)
            O[rbase + nt * 16 + l16] = o[nt][i] * inv;
    }
}

// ---------------------------------------------------------------------------
// Launch: 3 projection GEMMs -> bf16 workspace (48 MB, L2-resident), then
// TDM-fed flash attention. All on `stream`, graph-capture safe.
// ---------------------------------------------------------------------------
extern "C" void kernel_launch(void* const* d_in, const int* in_sizes, int n_in,
                              void* d_out, int out_size, void* d_ws, size_t ws_size,
                              hipStream_t stream) {
    (void)in_sizes; (void)n_in; (void)out_size; (void)ws_size;

    const float* q_in = (const float*)d_in[0];
    const float* k_in = (const float*)d_in[1];
    const float* v_in = (const float*)d_in[2];
    const float* Wq   = (const float*)d_in[3];
    const float* bq   = (const float*)d_in[4];
    const float* Wk   = (const float*)d_in[5];
    const float* bk   = (const float*)d_in[6];
    const float* Wv   = (const float*)d_in[7];
    const float* bv   = (const float*)d_in[8];

    const size_t elems = (size_t)BATCH * S_LEN * DMODEL;   // 8M elements each
    __bf16* Qw  = (__bf16*)d_ws;
    __bf16* Kw  = Qw + elems;
    __bf16* Vtw = Kw + elems;

    dim3 pgrid(BATCH * S_LEN / 128, DMODEL / 128);          // (128, 4)
    proj_gemm_bf16<<<pgrid, 256, 0, stream>>>(q_in, Wq, bq, Qw, 0);
    proj_gemm_bf16<<<pgrid, 256, 0, stream>>>(k_in, Wk, bk, Kw, 0);
    proj_gemm_bf16<<<pgrid, 256, 0, stream>>>(v_in, Wv, bv, Vtw, 1);

    dim3 agrid(S_LEN / 64, BATCH);                          // (64, 4)
    attn_fwd_bf16<<<agrid, 128, 0, stream>>>(Qw, Kw, Vtw, (float*)d_out);
}
